// PatchCrossAttention_82927228551672
// MI455X (gfx1250) — matrix operand order repaired and checked by reading, not compiled, not run on hardware
//
#include <hip/hip_runtime.h>

// Toggle: use gfx1250 async global->LDS staging (ASYNCcnt path). Set to 0 to fall
// back to synchronous global_load + ds_store staging.
#ifndef USE_ASYNC_LDS
#define USE_ASYNC_LDS 1
#endif

// ---------------- problem constants ----------------
constexpr int Bc = 4, Pc = 256, Sc = 4096, Dc = 1024, Hc = 16, HDc = 64;

typedef __attribute__((ext_vector_type(16))) __bf16          v16bf;
typedef __attribute__((ext_vector_type(16))) unsigned short  v16us;
typedef __attribute__((ext_vector_type(8)))  float           v8f;

__device__ __forceinline__ unsigned short f2bf(float f) {
    unsigned u = __float_as_uint(f);
    u += 0x7FFFu + ((u >> 16) & 1u);           // round-to-nearest-even
    return (unsigned short)(u >> 16);
}
__device__ __forceinline__ float bf2f(unsigned short h) {
    return __uint_as_float(((unsigned)h) << 16);
}

// swizzled position along K (groups of 32): [k0-7, k16-23, k8-15, k24-31]
// so an A-operand lane fragment (per CDNA5 ISA 16-bit A 16x32 layout) is contiguous.
__device__ __forceinline__ int swzA(int k) {
    const int g  = (k >> 3) & 3;
    const int pg = ((g & 1) << 1) | (g >> 1);   // 0->0, 1->2, 2->1, 3->3
    return (k & ~31) + pg * 8 + (k & 7);
}

union Frag32 { uint4 q[2]; v16bf v; };

__device__ __forceinline__ v8f wmma_bf16(const v16bf& a, const v16bf& b, v8f c) {
    return __builtin_amdgcn_wmma_f32_16x16x32_bf16(false, a, false, b, (short)0, c, false, false);
}

// async 16B global -> LDS copy per lane (GLOBAL_LOAD_ASYNC_TO_LDS_B128, ASYNCcnt)
__device__ __forceinline__ void stage16_async(const unsigned short* g, void* lds) {
#if USE_ASYNC_LDS
    const unsigned ldsAddr = (unsigned)(uintptr_t)lds;   // low 32 bits = LDS byte address
    asm volatile("global_load_async_to_lds_b128 %0, %1, off"
                 :: "v"(ldsAddr), "v"(g) : "memory");
#else
    *reinterpret_cast<uint4*>(lds) = *reinterpret_cast<const uint4*>(g);
#endif
}
__device__ __forceinline__ void stage_wait() {
#if USE_ASYNC_LDS
    asm volatile("s_wait_asynccnt 0x0" ::: "memory");
#endif
}

// split 16 fp32 into bf16 hi+lo fragments (bf16x3 scheme) -- fallback path only
__device__ __forceinline__ void cvt16(const float* f, v16bf& hi, v16bf& lo) {
    v16us uh, ul;
#pragma unroll
    for (int i = 0; i < 16; ++i) {
        unsigned short hb = f2bf(f[i]);
        uh[i] = hb;
        ul[i] = f2bf(f[i] - bf2f(hb));
    }
    hi = __builtin_bit_cast(v16bf, uh);
    lo = __builtin_bit_cast(v16bf, ul);
}

// ---------------- fp32 -> bf16 hi/lo split (optionally K-swizzled), streaming ----------------
template <bool SWZ>
__global__ __launch_bounds__(256) void cvt_split_kernel(
    const float* __restrict__ x, unsigned short* __restrict__ hi,
    unsigned short* __restrict__ lo, int n4)
{
    const int i = blockIdx.x * 256 + threadIdx.x;
    if (i >= n4) return;
    const int k = i * 4;
    const float4 v = reinterpret_cast<const float4*>(x)[i];
    ushort4 h, l;
    h.x = f2bf(v.x); l.x = f2bf(v.x - bf2f(h.x));
    h.y = f2bf(v.y); l.y = f2bf(v.y - bf2f(h.y));
    h.z = f2bf(v.z); l.z = f2bf(v.z - bf2f(h.z));
    h.w = f2bf(v.w); l.w = f2bf(v.w - bf2f(h.w));
    const int sk = SWZ ? swzA(k) : k;          // 4 consecutive elems never cross an 8-group
    *reinterpret_cast<ushort4*>(hi + sk) = h;
    *reinterpret_cast<ushort4*>(lo + sk) = l;
}

// ---------------- LayerNorm over D=1024, one row per 256-thread block ----------------
template <bool SPLIT>
__global__ __launch_bounds__(256) void ln1024_kernel(
    const float* __restrict__ x, const float* __restrict__ g,
    const float* __restrict__ bt, float* __restrict__ yf,
    unsigned short* __restrict__ yhi, unsigned short* __restrict__ ylo)
{
    __shared__ float red[8];
    const int row = blockIdx.x;
    const int t   = threadIdx.x;
    const float4 v = reinterpret_cast<const float4*>(x + (size_t)row * Dc)[t];

    float s = v.x + v.y + v.z + v.w;
#pragma unroll
    for (int o = 16; o > 0; o >>= 1) s += __shfl_xor(s, o, 32);
    if ((t & 31) == 0) red[t >> 5] = s;
    __syncthreads();
    float tot = red[0]+red[1]+red[2]+red[3]+red[4]+red[5]+red[6]+red[7];
    const float mean = tot * (1.0f / Dc);

    const float dx = v.x - mean, dy = v.y - mean, dz = v.z - mean, dw = v.w - mean;
    float s2 = dx*dx + dy*dy + dz*dz + dw*dw;
#pragma unroll
    for (int o = 16; o > 0; o >>= 1) s2 += __shfl_xor(s2, o, 32);
    __syncthreads();
    if ((t & 31) == 0) red[t >> 5] = s2;
    __syncthreads();
    const float var = (red[0]+red[1]+red[2]+red[3]+red[4]+red[5]+red[6]+red[7]) * (1.0f / Dc);
    const float rs  = rsqrtf(var + 1e-5f);

    const float4 gv = reinterpret_cast<const float4*>(g)[t];
    const float4 bv = reinterpret_cast<const float4*>(bt)[t];
    float4 o;
    o.x = dx * rs * gv.x + bv.x;
    o.y = dy * rs * gv.y + bv.y;
    o.z = dz * rs * gv.z + bv.z;
    o.w = dw * rs * gv.w + bv.w;

    if (SPLIT) {
        ushort4 h, l;
        h.x = f2bf(o.x); l.x = f2bf(o.x - bf2f(h.x));
        h.y = f2bf(o.y); l.y = f2bf(o.y - bf2f(h.y));
        h.z = f2bf(o.z); l.z = f2bf(o.z - bf2f(h.z));
        h.w = f2bf(o.w); l.w = f2bf(o.w - bf2f(h.w));
        const int sk = swzA(t * 4);
        *reinterpret_cast<ushort4*>(yhi + (size_t)row * Dc + sk) = h;
        *reinterpret_cast<ushort4*>(ylo + (size_t)row * Dc + sk) = l;
    } else {
        reinterpret_cast<float4*>(yf + (size_t)row * Dc)[t] = o;
    }
}

// ---------------- fast GEMM: pre-split bf16 operands, async-LDS-staged B, double buffered ----
// C = A (MxK) @ W^T + bias [+resid]; A in swizzled bf16 hi/lo, W in natural bf16 hi/lo.
// block = 256 threads = 8 waves; block tile 256x64; wave w -> rows [blk.x*256 + w*32, +32).
template <bool RESID>
__global__ __launch_bounds__(256) void gemm_pre_bf16x3(
    const unsigned short* __restrict__ Ahi, const unsigned short* __restrict__ Alo,
    const unsigned short* __restrict__ Whi, const unsigned short* __restrict__ Wlo,
    const float* __restrict__ bias, const float* __restrict__ resid,
    float* __restrict__ C, int M, int N, int K)
{
    // padded rows (40 ushorts = 80B): 16-lane ds_load_b128 hits 16 disjoint 4-bank groups
    __shared__ unsigned short wtile[2][2][64][40];     // 20 KB

    const int tid   = threadIdx.x;
    const int lane  = tid & 31;
    const int wave  = tid >> 5;
    const int l15   = lane & 15;
    const int kh    = lane >> 4;
    const int mBase = blockIdx.x * 256 + wave * 32;
    const int nBase = blockIdx.y * 64;

    // cooperative W staging: thread -> (row, 8-elem quarter)
    const int sRow = tid >> 2;
    const int sQ   = (tid & 3) * 8;
    const unsigned short* wsrcHi = Whi + (size_t)(nBase + sRow) * K + sQ;
    const unsigned short* wsrcLo = Wlo + (size_t)(nBase + sRow) * K + sQ;

    v8f acc[8] = {};
    const size_t aOff0 = (size_t)(mBase + l15) * K;
    const size_t aOff1 = (size_t)(mBase + 16 + l15) * K;

    // prologue: stage k-block 0 into buffer 0 (async path or sync fallback)
    stage16_async(wsrcHi, &wtile[0][0][sRow][sQ]);
    stage16_async(wsrcLo, &wtile[0][1][sRow][sQ]);
    stage_wait();
    __syncthreads();

    const int iters = K / 32;
    for (int it = 0; it < iters; ++it) {
        const int k0  = it * 32;
        const int buf = it & 1;
        const bool more = (it + 1) < iters;

#if USE_ASYNC_LDS
        // kick off DMA of the next W stage into the other buffer (last read in the previous iteration,
        // ordered by the previous end-of-iteration barrier)
        if (more) {
            stage16_async(wsrcHi + k0 + 32, &wtile[buf ^ 1][0][sRow][sQ]);
            stage16_async(wsrcLo + k0 + 32, &wtile[buf ^ 1][1][sRow][sQ]);
        }
#else
        uint4 nh, nl;
        if (more) {
            nh = *reinterpret_cast<const uint4*>(wsrcHi + k0 + 32);
            nl = *reinterpret_cast<const uint4*>(wsrcLo + k0 + 32);
        }
#endif

        // A fragments for both m-subtiles (swizzled layout -> contiguous 32B each)
        Frag32 a0h, a0l, a1h, a1l;
        {
            const uint4* p;
            p = reinterpret_cast<const uint4*>(Ahi + aOff0 + k0 + kh * 16); a0h.q[0]=p[0]; a0h.q[1]=p[1];
            p = reinterpret_cast<const uint4*>(Alo + aOff0 + k0 + kh * 16); a0l.q[0]=p[0]; a0l.q[1]=p[1];
            p = reinterpret_cast<const uint4*>(Ahi + aOff1 + k0 + kh * 16); a1h.q[0]=p[0]; a1h.q[1]=p[1];
            p = reinterpret_cast<const uint4*>(Alo + aOff1 + k0 + kh * 16); a1l.q[0]=p[0]; a1l.q[1]=p[1];
        }
        __builtin_prefetch(Ahi + aOff0 + k0 + 256, 0, 3);   // global_prefetch_b8
        __builtin_prefetch(Alo + aOff0 + k0 + 256, 0, 3);

#pragma unroll
        for (int tI = 0; tI < 4; ++tI) {
            Frag32 bh, bl;
            const uint4* q0 = reinterpret_cast<const uint4*>(&wtile[buf][0][tI * 16 + l15][kh * 8]);
            bh.q[0] = q0[0]; bh.q[1] = q0[2];
            const uint4* q1 = reinterpret_cast<const uint4*>(&wtile[buf][1][tI * 16 + l15][kh * 8]);
            bl.q[0] = q1[0]; bl.q[1] = q1[2];

            acc[tI]     = wmma_bf16(a0h.v, bh.v, acc[tI]);
            acc[tI]     = wmma_bf16(a0h.v, bl.v, acc[tI]);
            acc[tI]     = wmma_bf16(a0l.v, bh.v, acc[tI]);
            acc[4 + tI] = wmma_bf16(a1h.v, bh.v, acc[4 + tI]);
            acc[4 + tI] = wmma_bf16(a1h.v, bl.v, acc[4 + tI]);
            acc[4 + tI] = wmma_bf16(a1l.v, bh.v, acc[4 + tI]);
        }

#if USE_ASYNC_LDS
        stage_wait();                 // ASYNCcnt==0: DMA into buf^1 complete
#else
        if (more) {
            *reinterpret_cast<uint4*>(&wtile[buf ^ 1][0][sRow][sQ]) = nh;
            *reinterpret_cast<uint4*>(&wtile[buf ^ 1][1][sRow][sQ]) = nl;
        }
#endif
        __syncthreads();              // publish buf^1, retire reads of buf
    }

#pragma unroll
    for (int sTile = 0; sTile < 2; ++sTile) {
#pragma unroll
        for (int tI = 0; tI < 4; ++tI) {
            const int n  = nBase + tI * 16 + l15;
            const float bv = bias[n];
#pragma unroll
            for (int r = 0; r < 8; ++r) {
                const int m = mBase + sTile * 16 + kh * 8 + r;
                float val = acc[sTile * 4 + tI][r] + bv;
                if (RESID) val += resid[(size_t)m * N + n];
                C[(size_t)m * N + n] = val;
            }
        }
    }
}

// ---------------- fallback GEMM: fp32 operands, convert in-loop --------
template <bool RESID>
__global__ __launch_bounds__(256) void gemm_cvt_bf16x3(
    const float* __restrict__ A, const float* __restrict__ W,
    const float* __restrict__ bias, const float* __restrict__ resid,
    float* __restrict__ C, int M, int N, int K)
{
    const int lane  = threadIdx.x & 31;
    const int wave  = threadIdx.x >> 5;
    const int l15   = lane & 15;
    const int kh    = lane >> 4;
    const int mBase = blockIdx.x * 128 + wave * 16;
    const int nBase = blockIdx.y * 64;

    v8f acc[4] = {};
    const float* aRow = A + (size_t)(mBase + l15) * K;

    for (int k0 = 0; k0 < K; k0 += 32) {
        const float4* ap = reinterpret_cast<const float4*>(aRow + k0 + kh * 8);
        const float4 a0 = ap[0], a1 = ap[1], a2 = ap[4], a3 = ap[5];
        const float af[16] = {a0.x,a0.y,a0.z,a0.w, a1.x,a1.y,a1.z,a1.w,
                              a2.x,a2.y,a2.z,a2.w, a3.x,a3.y,a3.z,a3.w};
        v16bf ahi, alo;
        cvt16(af, ahi, alo);

#pragma unroll
        for (int tI = 0; tI < 4; ++tI) {
            const int nRow = nBase + tI * 16 + l15;
            const float4* wp = reinterpret_cast<const float4*>(W + (size_t)nRow * K + k0 + kh * 8);
            const float4 b0 = wp[0], b1 = wp[1], b2 = wp[4], b3 = wp[5];
            const float bf[16] = {b0.x,b0.y,b0.z,b0.w, b1.x,b1.y,b1.z,b1.w,
                                  b2.x,b2.y,b2.z,b2.w, b3.x,b3.y,b3.z,b3.w};
            v16bf bhi, blo;
            cvt16(bf, bhi, blo);

            acc[tI] = wmma_bf16(ahi, bhi, acc[tI]);
            acc[tI] = wmma_bf16(ahi, blo, acc[tI]);
            acc[tI] = wmma_bf16(alo, bhi, acc[tI]);
        }
    }

#pragma unroll
    for (int tI = 0; tI < 4; ++tI) {
        const int n  = nBase + tI * 16 + l15;
        const float bv = bias[n];
#pragma unroll
        for (int r = 0; r < 8; ++r) {
            const int m = mBase + kh * 8 + r;
            float val = acc[tI][r] + bv;
            if (RESID) val += resid[(size_t)m * N + n];
            C[(size_t)m * N + n] = val;
        }
    }
}

// ---------------- windowed attention: one block per (b,p); 16 lanes per head ----------------
template <bool SPLIT>
__global__ __launch_bounds__(256) void win_attn_kernel(
    const float* __restrict__ Q, const float* __restrict__ Kp, const float* __restrict__ Vp,
    const int* __restrict__ pos, const float* __restrict__ mask,
    float* __restrict__ OutF, unsigned short* __restrict__ Ohi, unsigned short* __restrict__ Olo)
{
    __shared__ float qs[Dc];
    __shared__ float sc[Hc * 256];

    const int bp  = blockIdx.x;
    const int b   = bp >> 8;                  // P = 256
    const int t   = threadIdx.x;
    const int h   = t >> 4;
    const int sub = t & 15;

    const int start = pos[2 * bp];
    const int end   = pos[2 * bp + 1];
    const bool valid = (start < Sc) && (end <= Sc) && (start < end);
    int L = valid ? (end - start) : 0;
    if (L > 256) L = 256;                     // harness window is 16

    reinterpret_cast<float4*>(qs)[t] = reinterpret_cast<const float4*>(Q + (size_t)bp * Dc)[t];
    __syncthreads();

    const float* qh = qs + h * HDc;
    for (int j = sub; j < L; j += 16) {
        const float* kr = Kp + ((size_t)b * Sc + start + j) * Dc + h * HDc;
        float s = 0.0f;
#pragma unroll
        for (int d4 = 0; d4 < HDc / 4; ++d4) {
            const float4 kq = reinterpret_cast<const float4*>(kr)[d4];
            s += qh[d4*4+0]*kq.x + qh[d4*4+1]*kq.y + qh[d4*4+2]*kq.z + qh[d4*4+3]*kq.w;
        }
        const bool allowed = (mask[(size_t)b * Sc + start + j] != 0.0f);
        sc[h * 256 + j] = allowed ? s * 0.125f : -1e9f;   // 1/sqrt(64)
    }
    __syncthreads();

    float mx = -1e30f;
    for (int j = sub; j < L; j += 16) mx = fmaxf(mx, sc[h * 256 + j]);
#pragma unroll
    for (int o = 8; o > 0; o >>= 1) mx = fmaxf(mx, __shfl_xor(mx, o, 16));

    float se = 0.0f;
    for (int j = sub; j < L; j += 16) {
        const float e = __expf(sc[h * 256 + j] - mx);
        sc[h * 256 + j] = e;
        se += e;
    }
#pragma unroll
    for (int o = 8; o > 0; o >>= 1) se += __shfl_xor(se, o, 16);
    const float inv = (se > 0.0f) ? (1.0f / se) : 0.0f;
    __syncthreads();

    float4 acc = {0.f, 0.f, 0.f, 0.f};
    for (int j = 0; j < L; ++j) {
        const float w = sc[h * 256 + j];
        const float4 vv = reinterpret_cast<const float4*>(
            Vp + ((size_t)b * Sc + start + j) * Dc + h * HDc)[sub];
        acc.x += w * vv.x; acc.y += w * vv.y; acc.z += w * vv.z; acc.w += w * vv.w;
    }
    acc.x *= inv; acc.y *= inv; acc.z *= inv; acc.w *= inv;

    if (SPLIT) {
        ushort4 hv, lv;
        hv.x = f2bf(acc.x); lv.x = f2bf(acc.x - bf2f(hv.x));
        hv.y = f2bf(acc.y); lv.y = f2bf(acc.y - bf2f(hv.y));
        hv.z = f2bf(acc.z); lv.z = f2bf(acc.z - bf2f(hv.z));
        hv.w = f2bf(acc.w); lv.w = f2bf(acc.w - bf2f(hv.w));
        const int sk = swzA(h * HDc + sub * 4);
        *reinterpret_cast<ushort4*>(Ohi + (size_t)bp * Dc + sk) = hv;
        *reinterpret_cast<ushort4*>(Olo + (size_t)bp * Dc + sk) = lv;
    } else {
        reinterpret_cast<float4*>(OutF + (size_t)bp * Dc + h * HDc)[sub] = acc;
    }
}

// ---------------- host launcher ----------------
extern "C" void kernel_launch(void* const* d_in, const int* in_sizes, int n_in,
                              void* d_out, int out_size, void* d_ws, size_t ws_size,
                              hipStream_t stream)
{
    (void)in_sizes; (void)n_in; (void)out_size;

    const float* patch_emb = (const float*)d_in[0];
    const float* byte_hs   = (const float*)d_in[1];
    const int*   pos       = (const int*)  d_in[2];
    const float* bmask     = (const float*)d_in[3];
    const float* ln_g      = (const float*)d_in[4];
    const float* ln_b      = (const float*)d_in[5];
    const float* w_q = (const float*)d_in[6];  const float* b_q = (const float*)d_in[7];
    const float* w_k = (const float*)d_in[8];  const float* b_k = (const float*)d_in[9];
    const float* w_v = (const float*)d_in[10]; const float* b_v = (const float*)d_in[11];
    const float* w_o = (const float*)d_in[12]; const float* b_o = (const float*)d_in[13];
    float* out = (float*)d_out;

    const size_t MP = (size_t)Bc * Pc;      // 1024
    const size_t MS = (size_t)Bc * Sc;      // 16384
    const size_t NB_X  = MS * Dc * 2;       // bf16 X buffer (33.5 MB)
    const size_t NB_W  = (size_t)Dc * Dc * 2;
    const size_t NB_QX = MP * Dc * 2;
    const size_t NB_QF = MP * Dc * 4;
    const size_t NB_KF = MS * Dc * 4;
    const size_t need_fast = 2*NB_X + 8*NB_W + 2*NB_QX + NB_QF + 2*NB_KF + 2*NB_QX;

    if (ws_size >= need_fast) {
        // -------- fast path: pre-split bf16 operands --------
        char* w = (char*)d_ws;
        unsigned short* Xhi  = (unsigned short*)w; w += NB_X;
        unsigned short* Xlo  = (unsigned short*)w; w += NB_X;
        unsigned short* Wb[8];                  // q.hi q.lo k.hi k.lo v.hi v.lo o.hi o.lo
        for (int i = 0; i < 8; ++i) { Wb[i] = (unsigned short*)w; w += NB_W; }
        unsigned short* QXhi = (unsigned short*)w; w += NB_QX;
        unsigned short* QXlo = (unsigned short*)w; w += NB_QX;
        unsigned short* AThi = (unsigned short*)w; w += NB_QX;
        unsigned short* ATlo = (unsigned short*)w; w += NB_QX;
        float* Qf = (float*)w; w += NB_QF;
        float* Kf = (float*)w; w += NB_KF;
        float* Vf = (float*)w; w += NB_KF;

        const int nX4 = (int)(MS * Dc / 4);
        const int nW4 = Dc * Dc / 4;
        cvt_split_kernel<true ><<<(nX4 + 255) / 256, 256, 0, stream>>>(byte_hs, Xhi, Xlo, nX4);
        cvt_split_kernel<false><<<(nW4 + 255) / 256, 256, 0, stream>>>(w_q, Wb[0], Wb[1], nW4);
        cvt_split_kernel<false><<<(nW4 + 255) / 256, 256, 0, stream>>>(w_k, Wb[2], Wb[3], nW4);
        cvt_split_kernel<false><<<(nW4 + 255) / 256, 256, 0, stream>>>(w_v, Wb[4], Wb[5], nW4);
        cvt_split_kernel<false><<<(nW4 + 255) / 256, 256, 0, stream>>>(w_o, Wb[6], Wb[7], nW4);

        ln1024_kernel<true><<<(int)MP, 256, 0, stream>>>(patch_emb, ln_g, ln_b,
                                                         nullptr, QXhi, QXlo);

        gemm_pre_bf16x3<false><<<dim3((int)MP/256, Dc/64), 256, 0, stream>>>(
            QXhi, QXlo, Wb[0], Wb[1], b_q, nullptr, Qf, (int)MP, Dc, Dc);
        gemm_pre_bf16x3<false><<<dim3((int)MS/256, Dc/64), 256, 0, stream>>>(
            Xhi, Xlo, Wb[2], Wb[3], b_k, nullptr, Kf, (int)MS, Dc, Dc);
        gemm_pre_bf16x3<false><<<dim3((int)MS/256, Dc/64), 256, 0, stream>>>(
            Xhi, Xlo, Wb[4], Wb[5], b_v, nullptr, Vf, (int)MS, Dc, Dc);

        win_attn_kernel<true><<<(int)MP, 256, 0, stream>>>(Qf, Kf, Vf, pos, bmask,
                                                           nullptr, AThi, ATlo);

        gemm_pre_bf16x3<true><<<dim3((int)MP/256, Dc/64), 256, 0, stream>>>(
            AThi, ATlo, Wb[6], Wb[7], b_o, patch_emb, out, (int)MP, Dc, Dc);
    } else {
        // -------- fallback path: fp32 operands, convert in-loop --------
        float* qx   = (float*)d_ws;
        float* Qf   = qx + MP * Dc;
        float* Kf   = Qf + MP * Dc;
        float* Vf   = Kf + MS * Dc;
        float* attn = Vf + MS * Dc;

        ln1024_kernel<false><<<(int)MP, 256, 0, stream>>>(patch_emb, ln_g, ln_b,
                                                          qx, nullptr, nullptr);
        gemm_cvt_bf16x3<false><<<dim3((int)MP/128, Dc/64), 256, 0, stream>>>(
            qx, w_q, b_q, nullptr, Qf, (int)MP, Dc, Dc);
        gemm_cvt_bf16x3<false><<<dim3((int)MS/128, Dc/64), 256, 0, stream>>>(
            byte_hs, w_k, b_k, nullptr, Kf, (int)MS, Dc, Dc);
        gemm_cvt_bf16x3<false><<<dim3((int)MS/128, Dc/64), 256, 0, stream>>>(
            byte_hs, w_v, b_v, nullptr, Vf, (int)MS, Dc, Dc);
        win_attn_kernel<false><<<(int)MP, 256, 0, stream>>>(Qf, Kf, Vf, pos, bmask,
                                                            attn, nullptr, nullptr);
        gemm_cvt_bf16x3<true><<<dim3((int)MP/128, Dc/64), 256, 0, stream>>>(
            attn, w_o, b_o, patch_emb, out, (int)MP, Dc, Dc);
    }
}
